// STN_52793738002732
// MI455X (gfx1250) — matrix operand order repaired and checked by reading, not compiled, not run on hardware
//
#include <hip/hip_runtime.h>
#include <math.h>

typedef __attribute__((ext_vector_type(16))) _Float16 v16h;
typedef __attribute__((ext_vector_type(8)))  float    v8f;
typedef __attribute__((ext_vector_type(4)))  unsigned int u32x4;
typedef __attribute__((ext_vector_type(8)))  int    i32x8;
typedef __attribute__((ext_vector_type(4)))  int    i32x4;

#define WMMA_F32_F16(A,B,C) __builtin_amdgcn_wmma_f32_16x16x32_f16(false,(A),false,(B),(short)0,(C),false,false)

#define NSAMP 16384
#define DIM   785

#if defined(__has_builtin)
#if __has_builtin(__builtin_amdgcn_tensor_load_to_lds) && __has_builtin(__builtin_amdgcn_s_wait_tensorcnt)
#define HAVE_TDM 1
#endif
#endif
#ifndef HAVE_TDM
#define HAVE_TDM 0
#endif

static __device__ __forceinline__ float leaky_(float v) { return v >= 0.f ? v : 0.2f * v; }

// Unconditional load from clamped address + value select (no exec-mask branches).
static __device__ __forceinline__ float ldg_sel(const float* __restrict__ p, int idx, int limit) {
  const float v = p[min(idx, limit - 1)];
  return idx < limit ? v : 0.f;
}

// ---------------------------------------------------------------------------
// Kernel 1: fused MLP head -> 4x4 symmetric eigh -> rotation params (tc, ts)
// One wave handles a 16-sample tile; all GEMMs on v_wmma_f32_16x16x32_f16.
// ---------------------------------------------------------------------------
__global__ __launch_bounds__(256) void stn_mlp_theta(
    const float* __restrict__ x,
    const float* __restrict__ W1, const float* __restrict__ b1,
    const float* __restrict__ W2, const float* __restrict__ b2,
    const float* __restrict__ g1, const float* __restrict__ be1,
    const float* __restrict__ m1, const float* __restrict__ v1,
    const float* __restrict__ W3, const float* __restrict__ b3,
    float* __restrict__ tc_out, float* __restrict__ ts_out)
{
  __shared__ float hbuf[8][16][32];
  const int wave = threadIdx.x >> 5;
  const int lane = threadIdx.x & 31;
  const int lh   = lane & 15;
  const bool hi  = lane >= 16;
  const int row0 = (blockIdx.x * 8 + wave) * 16;

  const float* xr  = x  + (size_t)(row0 + lh) * DIM;
  const float* w1a = W1 + (size_t)lh * DIM;
  const float* w1b = W1 + (size_t)(16 + lh) * DIM;

  const int kaOff = hi ? 16 : 0;
  const int kbOff = hi ? 16 : 0;

  // ---- GEMM1: y = x @ W1^T, K = 785 (24 full k-tiles + clamp/select tail) ----
  v8f c0 = {}; v8f c1 = {};
  for (int k0 = 0; k0 < 768; k0 += 32) {
    __builtin_prefetch(xr + k0 + 128, 0, 1);
    v16h a, bf0, bf1;
    #pragma unroll
    for (int j = 0; j < 16; ++j) {
      const int ka = k0 + kaOff + j;
      a[j] = (_Float16)xr[ka];
      const int kb = k0 + kbOff + j;
      bf0[j] = (_Float16)w1a[kb];
      bf1[j] = (_Float16)w1b[kb];
    }
    c0 = WMMA_F32_F16(a, bf0, c0);
    c1 = WMMA_F32_F16(a, bf1, c1);
  }
  { // tail: K = 768..784 valid, rest zero (branch-free)
    v16h a, bf0, bf1;
    #pragma unroll
    for (int j = 0; j < 16; ++j) {
      const int ka = 768 + kaOff + j;
      a[j] = (_Float16)ldg_sel(xr, ka, DIM);
      const int kb = 768 + kbOff + j;
      bf0[j] = (_Float16)ldg_sel(w1a, kb, DIM);
      bf1[j] = (_Float16)ldg_sel(w1b, kb, DIM);
    }
    c0 = WMMA_F32_F16(a, bf0, c0);
    c1 = WMMA_F32_F16(a, bf1, c1);
  }

  // h1 = leaky(y + b1); C layout: lane = column n, rows = (hi?8:0)+r
  {
    const float bb0 = b1[lh], bb1 = b1[16 + lh];
    #pragma unroll
    for (int r = 0; r < 8; ++r) {
      const int m = (hi ? 8 : 0) + r;
      hbuf[wave][m][lh]      = leaky_(c0[r] + bb0);
      hbuf[wave][m][16 + lh] = leaky_(c1[r] + bb1);
    }
  }
  __syncthreads();

  // ---- GEMM2: h2 = leaky(bn(h1 @ W2^T + b2)) ----
  v8f d0 = {}; v8f d1 = {};
  {
    v16h a, bf0, bf1;
    const float* w2a = W2 + lh * 32;
    const float* w2b = W2 + (16 + lh) * 32;
    #pragma unroll
    for (int j = 0; j < 16; ++j) {
      const int ka = kaOff + j;
      a[j] = (_Float16)hbuf[wave][lh][ka];
      const int kb = kbOff + j;
      bf0[j] = (_Float16)w2a[kb];
      bf1[j] = (_Float16)w2b[kb];
    }
    d0 = WMMA_F32_F16(a, bf0, d0);
    d1 = WMMA_F32_F16(a, bf1, d1);
  }
  __syncthreads();
  {
    const int n0 = lh, n1 = 16 + lh;
    const float sc0 = g1[n0] * rsqrtf(v1[n0] + 1e-5f);
    const float sc1 = g1[n1] * rsqrtf(v1[n1] + 1e-5f);
    const float sh0 = be1[n0] - sc0 * m1[n0];
    const float sh1 = be1[n1] - sc1 * m1[n1];
    const float bb0 = b2[n0], bb1 = b2[n1];
    #pragma unroll
    for (int r = 0; r < 8; ++r) {
      const int m = (hi ? 8 : 0) + r;
      hbuf[wave][m][n0] = leaky_(sc0 * (d0[r] + bb0) + sh0);
      hbuf[wave][m][n1] = leaky_(sc1 * (d1[r] + bb1) + sh1);
    }
  }
  __syncthreads();

  // ---- GEMM3: A_vec = h2 @ W3^T (10 valid columns, zero-padded, branch-free) ----
  v8f e0 = {};
  {
    v16h a, bf0;
    const bool okr = lh < 10;
    const float* w3r = W3 + (okr ? lh : 0) * 32;
    #pragma unroll
    for (int j = 0; j < 16; ++j) {
      const int ka = kaOff + j;
      a[j] = (_Float16)hbuf[wave][lh][ka];
      const int kb = kbOff + j;
      const float wv = w3r[kb];
      bf0[j] = (_Float16)(okr ? wv : 0.f);
    }
    e0 = WMMA_F32_F16(a, bf0, e0);
  }
  __syncthreads();
  {
    const float b3v = b3[lh < 10 ? lh : 0];
    const float bias = (lh < 10) ? b3v : 0.f;
    #pragma unroll
    for (int r = 0; r < 8; ++r) {
      const int m = (hi ? 8 : 0) + r;
      hbuf[wave][m][lh] = e0[r] + bias;
    }
  }
  __syncthreads();

  // ---- per-sample 4x4 symmetric eigh (cyclic Jacobi, register-resident) ----
  if (!hi) {
    const int m = lh;
    float av[10];
    #pragma unroll
    for (int j = 0; j < 10; ++j) av[j] = hbuf[wave][m][j];

    float Am[4][4], Vm[4][4];
    Am[0][0]=av[0]; Am[0][1]=av[1]; Am[0][2]=av[2]; Am[0][3]=av[3];
    Am[1][0]=av[1]; Am[1][1]=av[4]; Am[1][2]=av[5]; Am[1][3]=av[6];
    Am[2][0]=av[2]; Am[2][1]=av[5]; Am[2][2]=av[7]; Am[2][3]=av[8];
    Am[3][0]=av[3]; Am[3][1]=av[6]; Am[3][2]=av[8]; Am[3][3]=av[9];
    #pragma unroll
    for (int i = 0; i < 4; ++i)
      #pragma unroll
      for (int j2 = 0; j2 < 4; ++j2) Vm[i][j2] = (i == j2) ? 1.f : 0.f;

    #define JROT(P,Q) do {                                                       \
      float apq = Am[P][Q];                                                      \
      if (fabsf(apq) > 1e-12f) {                                                 \
        float th = (Am[Q][Q] - Am[P][P]) / (2.0f * apq);                         \
        float tt = (th >= 0.f ? 1.f : -1.f) / (fabsf(th) + sqrtf(1.f + th*th));  \
        float cc = rsqrtf(1.f + tt*tt);                                          \
        float ss = tt * cc;                                                      \
        _Pragma("unroll")                                                        \
        for (int i = 0; i < 4; ++i) { float u = Am[i][P], w = Am[i][Q];          \
          Am[i][P] = cc*u - ss*w; Am[i][Q] = ss*u + cc*w; }                      \
        _Pragma("unroll")                                                        \
        for (int i = 0; i < 4; ++i) { float u = Am[P][i], w = Am[Q][i];          \
          Am[P][i] = cc*u - ss*w; Am[Q][i] = ss*u + cc*w; }                      \
        _Pragma("unroll")                                                        \
        for (int i = 0; i < 4; ++i) { float u = Vm[i][P], w = Vm[i][Q];          \
          Vm[i][P] = cc*u - ss*w; Vm[i][Q] = ss*u + cc*w; }                      \
      } } while (0)

    for (int sweep = 0; sweep < 8; ++sweep) {
      JROT(0,1); JROT(0,2); JROT(0,3); JROT(1,2); JROT(1,3); JROT(2,3);
    }
    #undef JROT

    float tc = Vm[0][0], ts = Vm[1][0], best = Am[0][0];
    if (Am[1][1] < best) { best = Am[1][1]; tc = Vm[0][1]; ts = Vm[1][1]; }
    if (Am[2][2] < best) { best = Am[2][2]; tc = Vm[0][2]; ts = Vm[1][2]; }
    if (Am[3][3] < best) { best = Am[3][3]; tc = Vm[0][3]; ts = Vm[1][3]; }
    tc_out[row0 + m] = tc;
    ts_out[row0 + m] = ts;
  }
}

// ---------------------------------------------------------------------------
// Kernel 2: per-sample fused STN sampling + 4-layer conv net, all in LDS.
// Image staged via TDM (tensor_load_to_lds) when available.
// ---------------------------------------------------------------------------
__global__ __launch_bounds__(256) void stn_conv_net(
    const float* __restrict__ x,
    const float* __restrict__ tc_in, const float* __restrict__ ts_in,
    const float* __restrict__ c1w, const float* __restrict__ c1b,
    const float* __restrict__ g2, const float* __restrict__ be2,
    const float* __restrict__ m2, const float* __restrict__ v2,
    const float* __restrict__ c2w, const float* __restrict__ c2b,
    const float* __restrict__ g3, const float* __restrict__ be3,
    const float* __restrict__ m3, const float* __restrict__ v3,
    const float* __restrict__ c3w, const float* __restrict__ c3b,
    const float* __restrict__ g4, const float* __restrict__ be4,
    const float* __restrict__ m4, const float* __restrict__ v4,
    const float* __restrict__ c4w, const float* __restrict__ c4b,
    float* __restrict__ out)
{
  __shared__ float s_img[784];
  __shared__ float s_xi[784];
  __shared__ float s_z1[32][196];
  __shared__ float s_z2[32][49];
  __shared__ float s_z3[32][16];
  __shared__ float s_acc[10];

  const int tid  = threadIdx.x;
  const int samp = blockIdx.x;
  const int wave = tid >> 5;
  const int lane = tid & 31;
  const int lh   = lane & 15;
  const bool hi  = lane >= 16;

  const float* xrow = x + (size_t)samp * DIM;

  // ---- stage image into LDS (TDM 1-D tile when available) ----
#if HAVE_TDM
  if (wave == 0) {
    const unsigned long long ga = (unsigned long long)(uintptr_t)xrow;
    const unsigned lds_addr = (unsigned)(uintptr_t)(&s_img[0]);  // addr[31:0] = LDS offset
    u32x4 g0;
    g0.x = 1u;                                        // count=1, user descriptor
    g0.y = lds_addr;                                  // D#.lds_addr
    g0.z = (unsigned)(ga & 0xFFFFFFFFu);              // global_addr[31:0]
    g0.w = (unsigned)((ga >> 32) & 0x01FFFFFFu) | 0x80000000u;  // addr[56:32] | type=2
    i32x8 g1 = { 0x00020000,            // data_size = 4B, mask=0
                 (int)(784u << 16),     // tensor_dim0 = 784 (low half)
                 (int)(1u   << 16),     // tensor_dim1 = 1
                 (int)(784u << 16),     // tile_dim0 = 784
                 1,                     // tile_dim1 = 1
                 784,                   // tensor_dim0_stride = 784
                 0, 0 };
#if __has_include(<hip/amd_detail/amd_gfx1250_TDM.h>)
    __builtin_amdgcn_tensor_load_to_lds(g0, g1, (i32x4)0, (i32x4)0, (i32x8)0, 0);
#else
    __builtin_amdgcn_tensor_load_to_lds(g0, g1, (i32x4)0, (i32x4)0, 0);
#endif
    __builtin_amdgcn_s_wait_tensorcnt(0);
  }
#else
  for (int p = tid; p < 784; p += 256) s_img[p] = xrow[p];
#endif
  __syncthreads();

  // ---- affine grid + bilinear sample (clamp + select, branch-free) ----
  const float tc = tc_in[samp], ts = ts_in[samp];
  auto gat = [&](int yy, int xx) -> float {
    const bool ok = (xx >= 0) & (xx < 28) & (yy >= 0) & (yy < 28);
    const int yc = min(max(yy, 0), 27), xc = min(max(xx, 0), 27);
    const float v = s_img[yc * 28 + xc];
    return ok ? v : 0.f;
  };
  for (int p = tid; p < 784; p += 256) {
    const int oy = p / 28, ox = p % 28;
    const float X = (float)(ox * 2 + 1) * (1.f / 28.f) - 1.f;
    const float Y = (float)(oy * 2 + 1) * (1.f / 28.f) - 1.f;
    const float gx =  tc * X + ts * Y;
    const float gy = -ts * X + tc * Y;
    const float ix = ((gx + 1.f) * 28.f - 1.f) * 0.5f;
    const float iy = ((gy + 1.f) * 28.f - 1.f) * 0.5f;
    const float x0f = floorf(ix), y0f = floorf(iy);
    const float wx = ix - x0f, wy = iy - y0f;
    const int x0 = (int)x0f, y0 = (int)y0f;
    const float v00 = gat(y0,     x0);
    const float v01 = gat(y0,     x0 + 1);
    const float v10 = gat(y0 + 1, x0);
    const float v11 = gat(y0 + 1, x0 + 1);
    s_xi[p] = v00 * (1.f - wx) * (1.f - wy) + v01 * wx * (1.f - wy)
            + v10 * (1.f - wx) * wy         + v11 * wx * wy;
  }
  __syncthreads();

  // ---- conv1 (1->32, 3x3, s2, p1) + BN(g2) + ReLU, VALU, branch-free ----
  for (int o = tid; o < 32 * 196; o += 256) {
    const int c = o / 196, sp = o % 196;
    const int oy = sp / 14, ox = sp % 14;
    float sum = c1b[c];
    #pragma unroll
    for (int ky = 0; ky < 3; ++ky)
      #pragma unroll
      for (int kx = 0; kx < 3; ++kx) {
        const int iy = oy * 2 - 1 + ky, ixx = ox * 2 - 1 + kx;
        const bool ok = (iy >= 0) & (iy < 28) & (ixx >= 0) & (ixx < 28);
        const int iyc = min(max(iy, 0), 27), ixc = min(max(ixx, 0), 27);
        const float v = s_xi[iyc * 28 + ixc];
        sum += (ok ? v : 0.f) * c1w[c * 9 + ky * 3 + kx];
      }
    const float sc = g2[c] * rsqrtf(v2[c] + 1e-5f);
    s_z1[c][sp] = fmaxf(sc * (sum - m2[c]) + be2[c], 0.f);
  }
  __syncthreads();

  // ---- conv2 (32->32, 3x3, s2, p1) as WMMA im2col GEMM: M=32, N=49->64, K=288 ----
  {
    const int Mt = wave >> 2;
    const int Nt = wave & 3;
    const int n  = Nt * 16 + lh;
    const bool nok = n < 49;
    const int ncl = nok ? n : 48;
    const int oy = ncl / 7, ox = ncl % 7;
    const int mrow = Mt * 16 + lh;
    const float* wrow = c2w + mrow * 288;
    v8f acc = {};
    #pragma unroll
    for (int kt = 0; kt < 9; ++kt) {
      const int k0 = kt * 32;
      v16h a, bf;
      #pragma unroll
      for (int j = 0; j < 16; ++j) {
        const int ka = k0 + (hi ? 16 : 0) + j;
        a[j] = (_Float16)wrow[ka];
        const int kb = k0 + (hi ? 16 : 0) + j;     // constant after unroll
        const int ci = kb / 9, kk = kb % 9;
        const int ky = kk / 3, kx = kk % 3;
        const int iy = oy * 2 - 1 + ky, ixx = ox * 2 - 1 + kx;
        const bool ok = nok & (iy >= 0) & (iy < 14) & (ixx >= 0) & (ixx < 14);
        const int iyc = min(max(iy, 0), 13), ixc = min(max(ixx, 0), 13);
        const float v = s_z1[ci][iyc * 14 + ixc];
        bf[j] = (_Float16)(ok ? v : 0.f);
      }
      acc = WMMA_F32_F16(a, bf, acc);
    }
    if (nok) {
      #pragma unroll
      for (int r = 0; r < 8; ++r) {
        const int c = Mt * 16 + (hi ? 8 : 0) + r;
        const float sc = g3[c] * rsqrtf(v3[c] + 1e-5f);
        s_z2[c][n] = fmaxf(sc * (acc[r] + c2b[c] - m3[c]) + be3[c], 0.f);
      }
    }
  }
  __syncthreads();

  // ---- conv3 (32->32, 3x3, s2, p1) as WMMA: M=32, N=16 (4x4), K=288 ----
  if (wave < 2) {
    const int Mt = wave;
    const int n = lh;
    const int oy = n >> 2, ox = n & 3;
    const int mrow = Mt * 16 + lh;
    const float* wrow = c3w + mrow * 288;
    v8f acc = {};
    #pragma unroll
    for (int kt = 0; kt < 9; ++kt) {
      const int k0 = kt * 32;
      v16h a, bf;
      #pragma unroll
      for (int j = 0; j < 16; ++j) {
        const int ka = k0 + (hi ? 16 : 0) + j;
        a[j] = (_Float16)wrow[ka];
        const int kb = k0 + (hi ? 16 : 0) + j;
        const int ci = kb / 9, kk = kb % 9;
        const int ky = kk / 3, kx = kk % 3;
        const int iy = oy * 2 - 1 + ky, ixx = ox * 2 - 1 + kx;
        const bool ok = (iy >= 0) & (iy < 7) & (ixx >= 0) & (ixx < 7);
        const int iyc = min(max(iy, 0), 6), ixc = min(max(ixx, 0), 6);
        const float v = s_z2[ci][iyc * 7 + ixc];
        bf[j] = (_Float16)(ok ? v : 0.f);
      }
      acc = WMMA_F32_F16(a, bf, acc);
    }
    #pragma unroll
    for (int r = 0; r < 8; ++r) {
      const int c = Mt * 16 + (hi ? 8 : 0) + r;
      const float sc = g4[c] * rsqrtf(v4[c] + 1e-5f);
      s_z3[c][n] = fmaxf(sc * (acc[r] + c3b[c] - m4[c]) + be4[c], 0.f);
    }
  }
  __syncthreads();

  // ---- conv4 (32->10, 4x4) + ReLU; 1x1 spatial mean is identity ----
  if (tid < 10) s_acc[tid] = c4b[tid];
  __syncthreads();
  for (int idx = tid; idx < 320; idx += 256) {
    const int j = idx >> 5, c = idx & 31;
    float p = 0.f;
    #pragma unroll
    for (int t = 0; t < 16; ++t) p += s_z3[c][t] * c4w[j * 512 + c * 16 + t];
    atomicAdd(&s_acc[j], p);
  }
  __syncthreads();
  if (tid < 10) out[(size_t)samp * 10 + tid] = fmaxf(s_acc[tid], 0.f);
}

// ---------------------------------------------------------------------------
extern "C" void kernel_launch(void* const* d_in, const int* in_sizes, int n_in,
                              void* d_out, int out_size, void* d_ws, size_t ws_size,
                              hipStream_t stream) {
  const float* x   = (const float*)d_in[0];
  const float* W1  = (const float*)d_in[1];
  const float* b1  = (const float*)d_in[2];
  const float* W2  = (const float*)d_in[3];
  const float* b2  = (const float*)d_in[4];
  const float* g1  = (const float*)d_in[5];
  const float* be1 = (const float*)d_in[6];
  const float* m1  = (const float*)d_in[7];
  const float* v1  = (const float*)d_in[8];
  const float* W3  = (const float*)d_in[9];
  const float* b3  = (const float*)d_in[10];
  const float* c1w = (const float*)d_in[11];
  const float* c1b = (const float*)d_in[12];
  const float* g2  = (const float*)d_in[13];
  const float* be2 = (const float*)d_in[14];
  const float* m2  = (const float*)d_in[15];
  const float* v2  = (const float*)d_in[16];
  const float* c2w = (const float*)d_in[17];
  const float* c2b = (const float*)d_in[18];
  const float* g3  = (const float*)d_in[19];
  const float* be3 = (const float*)d_in[20];
  const float* m3  = (const float*)d_in[21];
  const float* v3  = (const float*)d_in[22];
  const float* c3w = (const float*)d_in[23];
  const float* c3b = (const float*)d_in[24];
  const float* g4  = (const float*)d_in[25];
  const float* be4 = (const float*)d_in[26];
  const float* m4  = (const float*)d_in[27];
  const float* v4  = (const float*)d_in[28];
  const float* c4w = (const float*)d_in[29];
  const float* c4b = (const float*)d_in[30];

  float* tc = (float*)d_ws;
  float* ts = tc + NSAMP;

  stn_mlp_theta<<<NSAMP / 128, 256, 0, stream>>>(
      x, W1, b1, W2, b2, g1, be1, m1, v1, W3, b3, tc, ts);

  stn_conv_net<<<NSAMP, 256, 0, stream>>>(
      x, tc, ts, c1w, c1b, g2, be2, m2, v2,
      c2w, c2b, g3, be3, m3, v3, c3w, c3b,
      g4, be4, m4, v4, c4w, c4b, (float*)d_out);
}
